// GraphSAGE_128849019131
// MI455X (gfx1250) — compile-verified, matrix operand order checked by reading, not run on hardware
//
#include <hip/hip_runtime.h>
#include <hip/hip_bf16.h>

// ---------------------------------------------------------------------------
// GraphSAGE 2-layer on MI455X (gfx1250, wave32, WMMA + TDM).
//   layer: out = x @ W_self + (segsum(x[src], dst)/max(deg,1)) @ W_neigh + b
// Edge aggregation: bandwidth-bound, wave32-per-edge f32 atomics (~3.3 GB).
// Dense part: v_wmma_f32_16x16x32_bf16; weights staged in LDS once per block
// via tensor_load_to_lds (TDM) + s_wait_tensorcnt; B-frags via ds_load_b128.
// ---------------------------------------------------------------------------

typedef __attribute__((ext_vector_type(16))) __bf16 v16bf;
typedef __attribute__((ext_vector_type(8)))  float  v8f;
typedef __attribute__((ext_vector_type(4)))  unsigned int u32x4;
typedef __attribute__((ext_vector_type(4)))  int i32x4;
typedef __attribute__((ext_vector_type(8)))  int i32x8;

#define DFEAT 128
#define WFRAG 16384               // bf16 elements per packed weight matrix
#define LAYER_W_ELEMS (2 * WFRAG) // self + neigh, contiguous

// ---------------------------------------------------------------------------
// deg[dst] += 1 per edge
// ---------------------------------------------------------------------------
__global__ void degree_kernel(const int* __restrict__ dst,
                              float* __restrict__ deg, int nE) {
    int i = blockIdx.x * blockDim.x + threadIdx.x;
    if (i < nE) atomicAdd(&deg[dst[i]], 1.0f);
}

// ---------------------------------------------------------------------------
// agg[dst[e], :] += x[src[e], :]   -- one wave32 per edge, float4 per lane
// ---------------------------------------------------------------------------
__global__ void scatter_kernel(const float* __restrict__ x,
                               const int* __restrict__ src,
                               const int* __restrict__ dst,
                               float* __restrict__ agg, int nE) {
    int e = blockIdx.x * (blockDim.x >> 5) + (threadIdx.x >> 5);
    int lane = threadIdx.x & 31;
    if (e >= nE) return;
    int s = src[e];
    int d = dst[e];
    const float4 v = ((const float4*)(x + (size_t)s * DFEAT))[lane];
    float* ap = agg + (size_t)d * DFEAT + lane * 4;
    atomicAdd(ap + 0, v.x);
    atomicAdd(ap + 1, v.y);
    atomicAdd(ap + 2, v.z);
    atomicAdd(ap + 3, v.w);
}

// ---------------------------------------------------------------------------
// Repack a 128x128 fp32 weight (k-major, out = x @ W) into WMMA bf16
// B-operand fragment order: wp[((kt*8 + nt)*32 + lane)*16 + i]
// (lane%16 -> N, lane/16 -> K-half; elem i: v=i/2,
//  k = kt*32 + (v&3)*2 + (i&1) + (lane>=16?8:0) + (v>>2)*16).
// ---------------------------------------------------------------------------
__global__ void pack_weights(const float* __restrict__ W,
                             __bf16* __restrict__ wp) {
    int idx = blockIdx.x * blockDim.x + threadIdx.x;   // 16384 total
    if (idx >= WFRAG) return;
    int i    = idx & 15;
    int lane = (idx >> 4) & 31;
    int nt   = (idx >> 9) & 7;
    int kt   = idx >> 12;
    int v = i >> 1;
    int k = kt * 32 + (v & 3) * 2 + (i & 1) + ((lane >> 4) ? 8 : 0) + (v >> 2) * 16;
    int n = nt * 16 + (lane & 15);
    wp[idx] = (__bf16)W[k * DFEAT + n];
}

// ---------------------------------------------------------------------------
// TDM: DMA the layer's packed weights (2 matrices = 64 KB bf16, contiguous)
// from global into LDS as one flat 1-D tile. D# per CDNA5 ISA:
//   group0: [1:0]=count=1 | lds_addr | global_addr[56:0] | type=2
//   group1: data_size=1 (2B), tensor_dim0=tile_dim0=32768, tile_dim1=1,
//           tensor_dim0_stride=32768
// ---------------------------------------------------------------------------
__device__ __forceinline__ void tdm_load_weights(const __bf16* __restrict__ gsrc,
                                                 __bf16* lds_dst) {
    unsigned long long ga = (unsigned long long)(size_t)gsrc;
    unsigned int       la = (unsigned int)(size_t)lds_dst;  // addr[31:0] = LDS offset
    const unsigned int nelem = (unsigned int)LAYER_W_ELEMS; // 32768 x 2B

    u32x4 g0;
    g0.x = 1u;                                             // count = 1 descriptor
    g0.y = la;                                             // lds_addr (bytes)
    g0.z = (unsigned int)(ga & 0xFFFFFFFFull);             // global_addr[31:0]
    g0.w = (unsigned int)((ga >> 32) & 0x1FFFFFFull)       // global_addr[56:32]
           | (2u << 30);                                   // type = 2 ("image")

    i32x8 g1;
    g1[0] = (int)(1u << 16);                               // data_size = 1 -> 2 bytes
    g1[1] = (int)((nelem & 0xFFFFu) << 16);                // tensor_dim0[15:0]
    g1[2] = (int)(((nelem >> 16) & 0xFFFFu)                // tensor_dim0[31:16]
                  | (1u << 16));                           // tensor_dim1 = 1
    g1[3] = (int)((nelem & 0xFFFFu) << 16);                // tile_dim0 = 32768
    g1[4] = 1;                                             // tile_dim1 = 1
    g1[5] = (int)nelem;                                    // tensor_dim0_stride[31:0]
    g1[6] = 0;
    g1[7] = 0;

    i32x4 gz = {0, 0, 0, 0};
#if __clang_major__ >= 23
    i32x8 gz8 = {0, 0, 0, 0, 0, 0, 0, 0};
    __builtin_amdgcn_tensor_load_to_lds(g0, g1, gz, gz, gz8, 0);
#else
    __builtin_amdgcn_tensor_load_to_lds(g0, g1, gz, gz, 0);
#endif
    __builtin_amdgcn_s_wait_tensorcnt(0);
}

// ---------------------------------------------------------------------------
// Build a 16-bit A-fragment: 16 scaled fp32 -> bf16 (RNE via v_cvt_pk).
// K offsets for element i: i<8 -> kb+i ; i>=8 -> kb+16+(i-8)
// ---------------------------------------------------------------------------
__device__ __forceinline__ v16bf load_frag_f32(const float* __restrict__ p,
                                               int kb, float s) {
    float4 a0 = *(const float4*)(p + kb);
    float4 a1 = *(const float4*)(p + kb + 4);
    float4 a2 = *(const float4*)(p + kb + 16);
    float4 a3 = *(const float4*)(p + kb + 20);
    v16bf r;
    r[0]  = (__bf16)(a0.x * s); r[1]  = (__bf16)(a0.y * s);
    r[2]  = (__bf16)(a0.z * s); r[3]  = (__bf16)(a0.w * s);
    r[4]  = (__bf16)(a1.x * s); r[5]  = (__bf16)(a1.y * s);
    r[6]  = (__bf16)(a1.z * s); r[7]  = (__bf16)(a1.w * s);
    r[8]  = (__bf16)(a2.x * s); r[9]  = (__bf16)(a2.y * s);
    r[10] = (__bf16)(a2.z * s); r[11] = (__bf16)(a2.w * s);
    r[12] = (__bf16)(a3.x * s); r[13] = (__bf16)(a3.y * s);
    r[14] = (__bf16)(a3.z * s); r[15] = (__bf16)(a3.w * s);
    return r;
}

// ---------------------------------------------------------------------------
// One wave per 16-node row block; 8 waves/block share LDS-staged weights.
// 4 ktiles x 8 ntiles x 2 operands = 64 v_wmma_f32_16x16x32_bf16 per wave.
// Accumulators start at 0 (inline-0 C for first WMMA); bias fused in epilogue.
// ---------------------------------------------------------------------------
__global__ __launch_bounds__(256) void
sage_wmma_layer(const float* __restrict__ x,
                const float* __restrict__ agg,
                const float* __restrict__ deg,
                const __bf16* __restrict__ Wpack,   // [Wself | Wneigh] frags
                const float* __restrict__ bias,
                float* __restrict__ out,
                int n_nodes, int do_relu) {
    __shared__ __bf16 lds_w[LAYER_W_ELEMS];   // 64 KB

    const int lane = threadIdx.x & 31;
    const int wave = threadIdx.x >> 5;
    const int rb   = blockIdx.x * (blockDim.x >> 5) + wave;
    const int node0 = rb * 16;
    const bool active = (node0 < n_nodes);

    // one TDM issue per block; TENSORcnt is per-wave, so issuer waits, then
    // the workgroup barrier makes the LDS tile visible to all waves.
    if (wave == 0) tdm_load_weights(Wpack, lds_w);
    __syncthreads();
    if (!active) return;

    const int m = lane & 15;
    int row = node0 + m;
    if (row >= n_nodes) row = n_nodes - 1;    // clamp (N % 16 == 0 in practice)
    const float* xrow = x   + (size_t)row * DFEAT;
    const float* arow = agg + (size_t)row * DFEAT;
    // fast v_rcp_f32 (1-ulp) instead of IEEE divide sequence: feeds a bf16
    // operand (8-bit mantissa), so full div_fixup precision is wasted cycles.
    const float  inv  = __builtin_amdgcn_rcpf(fmaxf(deg[row], 1.0f));
    const int khalf = (lane >> 4) * 8;

    v8f acc[8];
#pragma unroll
    for (int nt = 0; nt < 8; ++nt)
#pragma unroll
        for (int r = 0; r < 8; ++r) acc[nt][r] = 0.0f;

#pragma unroll
    for (int kt = 0; kt < 4; ++kt) {
        const int kb = kt * 32 + khalf;
        v16bf aS = load_frag_f32(xrow, kb, 1.0f);
        v16bf aN = load_frag_f32(arow, kb, inv);
#pragma unroll
        for (int nt = 0; nt < 8; ++nt) {
            const int fo = ((kt * 8 + nt) * 32 + lane) * 16;
            const v16bf bS = *(const v16bf*)(lds_w + fo);
            const v16bf bN = *(const v16bf*)(lds_w + WFRAG + fo);
            acc[nt] = __builtin_amdgcn_wmma_f32_16x16x32_bf16(
                false, aS, false, bS, (short)0, acc[nt], false, false);
            acc[nt] = __builtin_amdgcn_wmma_f32_16x16x32_bf16(
                false, aN, false, bN, (short)0, acc[nt], false, false);
        }
    }

    // D layout: element r -> M = r + 8*(lane/16), N = nt*16 + (lane%16)
#pragma unroll
    for (int nt = 0; nt < 8; ++nt) {
        const float bv = bias[nt * 16 + m];
#pragma unroll
        for (int r = 0; r < 8; ++r) {
            int M = r + 8 * (lane >> 4);
            int nrow = node0 + M;
            if (nrow < n_nodes) {
                float v = acc[nt][r] + bv;
                if (do_relu) v = fmaxf(v, 0.0f);
                out[(size_t)nrow * DFEAT + nt * 16 + m] = v;
            }
        }
    }
}

// ---------------------------------------------------------------------------
extern "C" void kernel_launch(void* const* d_in, const int* in_sizes, int n_in,
                              void* d_out, int out_size, void* d_ws, size_t ws_size,
                              hipStream_t stream) {
    const float* in_feat = (const float*)d_in[0];
    const float* W1s = (const float*)d_in[1];
    const float* W1n = (const float*)d_in[2];
    const float* b1  = (const float*)d_in[3];
    const float* W2s = (const float*)d_in[4];
    const float* W2n = (const float*)d_in[5];
    const float* b2  = (const float*)d_in[6];
    const int*   src = (const int*)d_in[7];
    const int*   dst = (const int*)d_in[8];

    const int nN = in_sizes[0] / DFEAT;
    const int nE = in_sizes[7];

    // workspace layout
    float*  deg = (float*)d_ws;                       // nN floats
    float*  agg = deg + nN;                           // nN*128 floats
    float*  h   = agg + (size_t)nN * DFEAT;           // nN*128 floats
    __bf16* wp1 = (__bf16*)(h + (size_t)nN * DFEAT);  // layer1: [self|neigh]
    __bf16* wp2 = wp1 + LAYER_W_ELEMS;                // layer2: [self|neigh]

    // zero deg + agg (contiguous)
    hipMemsetAsync(deg, 0, ((size_t)nN + (size_t)nN * DFEAT) * sizeof(float), stream);

    // repack weights into WMMA bf16 B-fragment order (64 KB/layer, one-shot)
    pack_weights<<<64, 256, 0, stream>>>(W1s, wp1);
    pack_weights<<<64, 256, 0, stream>>>(W1n, wp1 + WFRAG);
    pack_weights<<<64, 256, 0, stream>>>(W2s, wp2);
    pack_weights<<<64, 256, 0, stream>>>(W2n, wp2 + WFRAG);

    // degree (shared by both layers)
    degree_kernel<<<(nE + 255) / 256, 256, 0, stream>>>(dst, deg, nE);

    const int rb = (nN + 15) / 16;          // 16-node row blocks
    const int blocks = (rb + 7) / 8;        // 8 waves per block

    // ---- layer 1 ----
    scatter_kernel<<<(nE + 7) / 8, 256, 0, stream>>>(in_feat, src, dst, agg, nE);
    sage_wmma_layer<<<blocks, 256, 0, stream>>>(
        in_feat, agg, deg, wp1, b1, h, nN, /*relu=*/1);

    // ---- layer 2 ----
    hipMemsetAsync(agg, 0, (size_t)nN * DFEAT * sizeof(float), stream);
    scatter_kernel<<<(nE + 7) / 8, 256, 0, stream>>>(h, src, dst, agg, nE);
    sage_wmma_layer<<<blocks, 256, 0, stream>>>(
        h, agg, deg, wp2, b2, (float*)d_out, nN, /*relu=*/0);
}
